// MultiHeadAttention_63075889709611
// MI455X (gfx1250) — matrix operand order repaired and checked by reading, not compiled, not run on hardware
//
#include <hip/hip_runtime.h>

// ---------------- problem constants (match reference) ----------------
#define H_DIM  2048
#define HQ     16
#define HKV    4
#define DH     128
#define REP    (HQ / HKV)
#define QSCALE 0.088388347648318447f  // 1/sqrt(128)

typedef __bf16 bf16;
typedef __attribute__((ext_vector_type(8)))  float  v8f;
typedef __attribute__((ext_vector_type(16))) __bf16 v16bf;
typedef __attribute__((ext_vector_type(8)))  __bf16 v8bf;

static __device__ __forceinline__ v8f v8f_zero() {
  v8f z = {0.f, 0.f, 0.f, 0.f, 0.f, 0.f, 0.f, 0.f};
  return z;
}

// round-to-nearest-even f32 -> bf16
static __device__ __forceinline__ bf16 f2bf(float f) {
  unsigned int u = __float_as_uint(f);
  u += 0x7fffu + ((u >> 16) & 1u);
  unsigned short s = (unsigned short)(u >> 16);
  bf16 b;
  __builtin_memcpy(&b, &s, 2);
  return b;
}

static __device__ __forceinline__ v8f wmma_bf16(v16bf a, v16bf b, v8f c) {
  return __builtin_amdgcn_wmma_f32_16x16x32_bf16(false, a, false, b, (short)0, c,
                                                 false, false);
}

// CDNA5 async global->LDS copy (16B per lane), tracked by ASYNCcnt.
static __device__ __forceinline__ void async_copy_b128(unsigned lds_byte_addr,
                                                       const bf16* gaddr) {
  asm volatile("global_load_async_to_lds_b128 %0, %1, off"
               :: "v"(lds_byte_addr), "v"(gaddr)
               : "memory");
}
#define WAIT_ASYNC(n) asm volatile("s_wait_asynccnt " #n ::: "memory")

static __device__ __forceinline__ unsigned lds_addr_of(const void* p) {
  return (unsigned)(unsigned long long)p;  // low 32 bits of shared-aperture addr
}

// A-matrix fragment (16x32), from row-major [M][K] LDS tile.
static __device__ __forceinline__ v16bf load_frag_a(const bf16* base, int stride, int lane) {
  const int hi = lane >> 4, r = lane & 15;
  const bf16* p = base + r * stride + 8 * hi;
  union { v16bf v; v8bf h[2]; } u;
  u.h[0] = *(const v8bf*)p;
  u.h[1] = *(const v8bf*)(p + 16);
  return u.v;
}

// B-matrix fragment (32x16 KxN), stored as row-major [N][K] LDS tile.
static __device__ __forceinline__ v16bf load_frag_b(const bf16* base, int stride, int lane) {
  const int hi = lane >> 4, r = lane & 15;
  const bf16* p = base + r * stride + 8 * hi;
  union { v16bf v; v8bf h[2]; } u;
  u.h[0] = *(const v8bf*)p;
  u.h[1] = *(const v8bf*)(p + 16);
  return u.v;
}

// ---------------- RMSNorm: x[row][H] -> bf16 hb[row][H] ----------------
__global__ __launch_bounds__(256) void rmsnorm_kernel(const float* __restrict__ x,
                                                      const float* __restrict__ w,
                                                      bf16* __restrict__ hb) {
  const int row = blockIdx.x;
  const float* xr = x + (size_t)row * H_DIM;
  float ss = 0.f;
  for (int i = threadIdx.x; i < H_DIM; i += 256) { float v = xr[i]; ss += v * v; }
  __shared__ float red[256];
  red[threadIdx.x] = ss;
  __syncthreads();
  for (int s = 128; s > 0; s >>= 1) {
    if (threadIdx.x < (unsigned)s) red[threadIdx.x] += red[threadIdx.x + s];
    __syncthreads();
  }
  const float scale = rsqrtf(red[0] * (1.0f / H_DIM) + 1e-6f);
  bf16* out = hb + (size_t)row * H_DIM;
  for (int i = threadIdx.x; i < H_DIM; i += 256) out[i] = f2bf(xr[i] * scale * w[i]);
}

// ---------------- fp32 -> bf16 convert ----------------
__global__ __launch_bounds__(256) void cvt_kernel(const float* __restrict__ in,
                                                  bf16* __restrict__ out, int n) {
  int i = blockIdx.x * 256 + threadIdx.x;
  if (i < n) out[i] = f2bf(in[i]);
}

// ---------------- GEMM: C[M][N] = A[M][K] * W[N][K]^T (+bias) ----------------
// 128x128 block tile, BK=32, quad-buffered async LDS pipeline (2 tiles in
// flight), single barrier per K-step.
#define GBM 128
#define GBN 128
#define GBK 32
#define GSTR (GBK + 8)              // 40 bf16 -> conflict-free frag loads
#define GBUF_B (GBM * GSTR * 2u)    // 10240 bytes per buffer
#define GEMM_SMEM (8u * GBUF_B)     // 4 A buffers + 4 W buffers = 81920 B

__global__ __launch_bounds__(256) void gemm_kernel(const bf16* __restrict__ A,
                                                   const bf16* __restrict__ W,
                                                   const float* __restrict__ bias,
                                                   float* __restrict__ C,
                                                   int M, int N, int K) {
  extern __shared__ __align__(16) char gsm[];
  const int tid = threadIdx.x;
  const int lane = tid & 31;
  const int wave = tid >> 5;       // 8 waves
  const int wm = wave & 3;         // 4 row groups of 32
  const int wn = wave >> 2;        // 2 col groups of 64
  const int m0 = blockIdx.y * GBM;
  const int n0 = blockIdx.x * GBN;

  const unsigned ldsBase = lds_addr_of(gsm);

  // per-thread load coords: 256 threads cover 64 rows x 4 segs; rows r and r+64
  const int lr = tid >> 2;         // 0..63
  const int lsg = tid & 3;         // 16B segment within a 32-elem row
  const bf16* Ag = A + (size_t)(m0 + lr) * K + lsg * 8;
  const bf16* Wg = W + (size_t)(n0 + lr) * K + lsg * 8;
  const unsigned ldOff = (unsigned)((lr * GSTR + lsg * 8) * 2);

  auto issue = [&](int t, int buf) {
    const bf16* a = Ag + (size_t)t * GBK;
    const bf16* w = Wg + (size_t)t * GBK;
    const unsigned ab = ldsBase + (unsigned)buf * GBUF_B;
    const unsigned wb = ldsBase + 4u * GBUF_B + (unsigned)buf * GBUF_B;
    async_copy_b128(ab + ldOff, a);
    async_copy_b128(ab + ldOff + 64u * GSTR * 2, a + (size_t)64 * K);
    async_copy_b128(wb + ldOff, w);
    async_copy_b128(wb + ldOff + 64u * GSTR * 2, w + (size_t)64 * K);
  };

  v8f acc[2][4];
  for (int i = 0; i < 2; i++)
    for (int j = 0; j < 4; j++) acc[i][j] = v8f_zero();

  const int nt = K / GBK;          // 64 for K=2048
  issue(0, 0);
  if (nt > 1) issue(1, 1);
  for (int t = 0; t < nt; ++t) {
    if (t + 2 < nt) {
      issue(t + 2, (t + 2) & 3);
      WAIT_ASYNC(0x8);             // tiles t+1, t+2 in flight; tile t drained
    } else if (t + 1 < nt) {
      WAIT_ASYNC(0x4);
    } else {
      WAIT_ASYNC(0x0);
    }
    __syncthreads();               // single barrier per step (quad buffer)

    const bf16* Ab = (const bf16*)(gsm + (unsigned)(t & 3) * GBUF_B);
    const bf16* Wb = (const bf16*)(gsm + 4u * GBUF_B + (unsigned)(t & 3) * GBUF_B);
    v16bf a0 = load_frag_a(Ab + (wm * 32 + 0) * GSTR, GSTR, lane);
    v16bf a1 = load_frag_a(Ab + (wm * 32 + 16) * GSTR, GSTR, lane);
    for (int j = 0; j < 4; j++) {
      v16bf b = load_frag_b(Wb + (wn * 64 + j * 16) * GSTR, GSTR, lane);
      acc[0][j] = wmma_bf16(a0, b, acc[0][j]);
      acc[1][j] = wmma_bf16(a1, b, acc[1][j]);
    }
  }

  const int hi = lane >> 4, ln = lane & 15;
  for (int i = 0; i < 2; i++)
    for (int j = 0; j < 4; j++) {
      const int col = n0 + wn * 64 + j * 16 + ln;
      const float bv = bias ? bias[col] : 0.f;
      for (int r = 0; r < 8; r++) {
        const int row = m0 + wm * 32 + i * 16 + r + 8 * hi;
        C[(size_t)row * N + col] = acc[i][j][r] + bv;
      }
    }
}

// ---------------- RoPE + layout ----------------
__global__ __launch_bounds__(256) void rope_q_kernel(const float* __restrict__ qlin,
                                                     bf16* __restrict__ qb, int S) {
  const int row = blockIdx.x;              // b*S + s
  const int b = row / S, s = row % S;
  const float* src = qlin + (size_t)row * (HQ * DH);
  const float pos = (float)s;
  for (int p = threadIdx.x; p < HQ * (DH / 2); p += 256) {
    const int h = p >> 6;
    const int i = p & 63;
    const float freq = __powf(10000.f, -((float)(2 * i)) * (1.f / DH));
    float sn, cs;
    __sincosf(pos * freq, &sn, &cs);
    const float t1 = src[h * DH + i];
    const float t2 = src[h * DH + i + 64];
    const size_t base = (((size_t)b * HQ + h) * S + s) * DH;
    qb[base + i]      = f2bf((t1 * cs - t2 * sn) * QSCALE);
    qb[base + i + 64] = f2bf((t2 * cs + t1 * sn) * QSCALE);
  }
}

__global__ __launch_bounds__(256) void rope_k_kernel(const float* __restrict__ klin,
                                                     bf16* __restrict__ kbuf, int S) {
  const int row = blockIdx.x;
  const int b = row / S, s = row % S;
  const float* src = klin + (size_t)row * (HKV * DH);
  const float pos = (float)s;
  for (int p = threadIdx.x; p < HKV * (DH / 2); p += 256) {
    const int h = p >> 6;
    const int i = p & 63;
    const float freq = __powf(10000.f, -((float)(2 * i)) * (1.f / DH));
    float sn, cs;
    __sincosf(pos * freq, &sn, &cs);
    const float t1 = src[h * DH + i];
    const float t2 = src[h * DH + i + 64];
    const size_t base = (((size_t)b * HKV + h) * S + s) * DH;
    kbuf[base + i]      = f2bf(t1 * cs - t2 * sn);
    kbuf[base + i + 64] = f2bf(t2 * cs + t1 * sn);
  }
}

// v_lin[B*S][HKV*DH] -> vt[B][HKV][DH][S] bf16 (transposed so P*V is A*W^T form)
__global__ __launch_bounds__(256) void vtrans_kernel(const float* __restrict__ vlin,
                                                     bf16* __restrict__ vt, int S) {
  const int row = blockIdx.x;
  const int b = row / S, s = row % S;
  const float* src = vlin + (size_t)row * (HKV * DH);
  for (int c = threadIdx.x; c < HKV * DH; c += 256) {
    const int h = c >> 7, d = c & 127;
    vt[(((size_t)b * HKV + h) * DH + d) * S + s] = f2bf(src[c]);
  }
}

// ---------------- Flash attention: 64-row Q tile per block, 4 waves ----------------
// Dynamic LDS layout (bytes): [Qs 17408][Ps 9216][Ks0 17408][Ks1 17408][Vs0 18432][Vs1 18432]
#define QK_STR (DH + 8)       // 136
#define VT_STR (64 + 8)       // 72
#define PS_STR (64 + 8)       // 72
#define QS_B   (64u * QK_STR * 2u)       // 17408
#define PS_B   (4u * 16u * PS_STR * 2u)  // 9216
#define KS_B   (64u * QK_STR * 2u)       // 17408
#define VS_B   ((unsigned)DH * VT_STR * 2u)  // 18432
#define OFF_PS (QS_B)
#define OFF_KS (QS_B + PS_B)
#define OFF_VS (QS_B + PS_B + 2u * KS_B)
#define ATTN_SMEM (QS_B + PS_B + 2u * KS_B + 2u * VS_B)  // 98304 B

__global__ __launch_bounds__(128) void attn_kernel(const bf16* __restrict__ qb,
                                                   const bf16* __restrict__ kb,
                                                   const bf16* __restrict__ vt,
                                                   bf16* __restrict__ ob, int S) {
  extern __shared__ __align__(16) char asm_lds[];
  const int qt = blockIdx.x;       // query tile (64 rows)
  const int h  = blockIdx.y;       // q head
  const int b  = blockIdx.z;
  const int g  = h / REP;          // kv head
  const int tid = threadIdx.x;
  const int lane = tid & 31;
  const int wave = tid >> 5;       // 4 waves x 16 query rows
  const int hi = lane >> 4, ln = lane & 15;

  const unsigned ldsBase = lds_addr_of(asm_lds);
  const bf16* Qs = (const bf16*)(asm_lds);
  bf16*       Ps = (bf16*)(asm_lds + OFF_PS);

  const bf16* ksrc = kb + (((size_t)b * HKV + g) * S) * DH;
  const bf16* vsrc = vt + (((size_t)b * HKV + g) * DH) * S;

  // Q tile [64][128] via async copy (overlaps with KV tile 0)
  const bf16* qsrc = qb + (((size_t)b * HQ + h) * S + (size_t)qt * 64) * DH;
  for (int i = 0; i < 8; i++) {
    const int u = tid + i * 128;
    const int r = u >> 4, sg = u & 15;
    async_copy_b128(ldsBase + (unsigned)((r * QK_STR + sg * 8) * 2),
                    qsrc + (size_t)r * DH + sg * 8);
  }

  auto issue_kv = [&](int j, int buf) {
    const unsigned kB = ldsBase + OFF_KS + (unsigned)buf * KS_B;
    const unsigned vB = ldsBase + OFF_VS + (unsigned)buf * VS_B;
    for (int i = 0; i < 8; i++) {
      const int u = tid + i * 128;
      const int r = u >> 4, sg = u & 15;
      async_copy_b128(kB + (unsigned)((r * QK_STR + sg * 8) * 2),
                      ksrc + ((size_t)(j * 64 + r)) * DH + sg * 8);
    }
    for (int i = 0; i < 8; i++) {
      const int u = tid + i * 128;
      const int r = u >> 3, sg = u & 7;
      async_copy_b128(vB + (unsigned)((r * VT_STR + sg * 8) * 2),
                      vsrc + (size_t)r * S + j * 64 + sg * 8);
    }
  };

  issue_kv(0, 0);
  WAIT_ASYNC(0x10);   // drain Q's 8 ops (in-order); KV0's 16 may stay in flight
  __syncthreads();

  // hoist Q A-fragments (fixed for all KV steps)
  v16bf qa[4];
  for (int c = 0; c < 4; c++)
    qa[c] = load_frag_a(Qs + wave * 16 * QK_STR + c * 32, QK_STR, lane);

  v8f o[8];
  float m_i[8], l_i[8];
  for (int n = 0; n < 8; n++) o[n] = v8f_zero();
  for (int r = 0; r < 8; r++) { m_i[r] = -__builtin_inff(); l_i[r] = 0.f; }

  for (int j = 0; j <= qt; ++j) {
    if (j < qt) {
      issue_kv(j + 1, (j + 1) & 1);
      WAIT_ASYNC(0x10);            // current tile drained, next in flight
    } else {
      WAIT_ASYNC(0x0);
    }
    __syncthreads();

    const bf16* Kp = (const bf16*)(asm_lds + OFF_KS + (unsigned)(j & 1) * KS_B);
    const bf16* Vp = (const bf16*)(asm_lds + OFF_VS + (unsigned)(j & 1) * VS_B);

    // scores S = Q K^T  (16 rows x 64 keys per wave)
    v8f sc[4];
    for (int t = 0; t < 4; t++) {
      v8f acc = v8f_zero();
      for (int c = 0; c < 4; c++) {
        v16bf bb = load_frag_b(Kp + t * 16 * QK_STR + c * 32, QK_STR, lane);
        acc = wmma_bf16(qa[c], bb, acc);
      }
      sc[t] = acc;
    }

    // causal mask (diagonal tile only)
    if (j == qt) {
      for (int t = 0; t < 4; t++)
        for (int r = 0; r < 8; r++) {
          const int key = j * 64 + t * 16 + ln;
          const int qr  = qt * 64 + wave * 16 + r + 8 * hi;
          if (key > qr) sc[t][r] = -__builtin_inff();
        }
    }

    // online softmax (row reductions across the 16 lanes of each half-wave)
    float alpha[8];
    for (int r = 0; r < 8; r++) {
      float v = fmaxf(fmaxf(sc[0][r], sc[1][r]), fmaxf(sc[2][r], sc[3][r]));
      for (int off = 1; off < 16; off <<= 1) v = fmaxf(v, __shfl_xor(v, off, 32));
      const float mnew = fmaxf(m_i[r], v);
      alpha[r] = __expf(m_i[r] - mnew);
      m_i[r] = mnew;
    }
    for (int r = 0; r < 8; r++) {
      float s0 = 0.f;
      for (int t = 0; t < 4; t++) {
        const float p = __expf(sc[t][r] - m_i[r]);
        sc[t][r] = p;
        s0 += p;
      }
      for (int off = 1; off < 16; off <<= 1) s0 += __shfl_xor(s0, off, 32);
      l_i[r] = l_i[r] * alpha[r] + s0;
    }
    for (int n = 0; n < 8; n++)
      for (int r = 0; r < 8; r++) o[n][r] *= alpha[r];

    // stage P (bf16) through per-wave LDS: C-layout -> A-fragment layout
    bf16* ps = Ps + wave * 16 * PS_STR;
    for (int t = 0; t < 4; t++)
      for (int r = 0; r < 8; r++)
        ps[(r + 8 * hi) * PS_STR + t * 16 + ln] = f2bf(sc[t][r]);
    __asm__ volatile("s_wait_dscnt 0" ::: "memory");

    // O += P * V   (V stored transposed: B-frag rows are d-dims)
    for (int c = 0; c < 2; c++) {
      v16bf pa = load_frag_a(ps + c * 32, PS_STR, lane);
      for (int n = 0; n < 8; n++) {
        v16bf vb = load_frag_b(Vp + n * 16 * VT_STR + c * 32, VT_STR, lane);
        o[n] = wmma_bf16(pa, vb, o[n]);
      }
    }
    __syncthreads();   // protects K/V buffer reuse (double buffer, 2 bars/step)
  }

  // normalize; write [B][S][HQ*DH] bf16 (feeds O-projection GEMM)
  for (int n = 0; n < 8; n++)
    for (int r = 0; r < 8; r++) {
      const int srow = qt * 64 + wave * 16 + r + 8 * hi;
      const int col  = h * DH + n * 16 + ln;
      const float val = o[n][r] / l_i[r];
      ob[((size_t)b * S + srow) * (size_t)(HQ * DH) + col] = f2bf(val);
    }
}

// ---------------- host side ----------------
extern "C" void kernel_launch(void* const* d_in, const int* in_sizes, int n_in,
                              void* d_out, int out_size, void* d_ws, size_t ws_size,
                              hipStream_t stream) {
  (void)n_in; (void)out_size; (void)ws_size;
  const float* x      = (const float*)d_in[0];
  const float* wq_f   = (const float*)d_in[1];
  const float* wk_f   = (const float*)d_in[2];
  const float* wv_f   = (const float*)d_in[3];
  const float* q_bias = (const float*)d_in[4];
  const float* k_bias = (const float*)d_in[5];
  const float* v_bias = (const float*)d_in[6];
  const float* wo_f   = (const float*)d_in[7];
  const float* ln_w   = (const float*)d_in[8];

  const int S = 2048;
  const int B = in_sizes[0] / (S * H_DIM);   // = 2
  const int M = B * S;

  char* ws = (char*)d_ws;
  size_t off = 0;
  auto take = [&](size_t bytes) -> void* {
    size_t cur = (off + 255) & ~(size_t)255;
    off = cur + bytes;
    return (void*)(ws + cur);
  };

  bf16*  hb    = (bf16*)take((size_t)M * H_DIM * 2);
  bf16*  wq    = (bf16*)take((size_t)HQ  * DH * H_DIM * 2);
  bf16*  wk    = (bf16*)take((size_t)HKV * DH * H_DIM * 2);
  bf16*  wv    = (bf16*)take((size_t)HKV * DH * H_DIM * 2);
  bf16*  wo    = (bf16*)take((size_t)H_DIM * HQ * DH * 2);
  float* q_lin = (float*)take((size_t)M * HQ  * DH * 4);
  float* k_lin = (float*)take((size_t)M * HKV * DH * 4);
  float* v_lin = (float*)take((size_t)M * HKV * DH * 4);
  bf16*  qbuf  = (bf16*)take((size_t)B * HQ  * S * DH * 2);
  bf16*  kbuf  = (bf16*)take((size_t)B * HKV * S * DH * 2);
  bf16*  vtb   = (bf16*)take((size_t)B * HKV * DH * S * 2);
  bf16*  attn  = hb;  // hb is dead once the QKV GEMMs complete

  rmsnorm_kernel<<<M, 256, 0, stream>>>(x, ln_w, hb);

  {
    int nq = HQ * DH * H_DIM, nk = HKV * DH * H_DIM, no = H_DIM * HQ * DH;
    cvt_kernel<<<(nq + 255) / 256, 256, 0, stream>>>(wq_f, wq, nq);
    cvt_kernel<<<(nk + 255) / 256, 256, 0, stream>>>(wk_f, wk, nk);
    cvt_kernel<<<(nk + 255) / 256, 256, 0, stream>>>(wv_f, wv, nk);
    cvt_kernel<<<(no + 255) / 256, 256, 0, stream>>>(wo_f, wo, no);
  }

  gemm_kernel<<<dim3((HQ * DH) / GBN, M / GBM), 256, GEMM_SMEM, stream>>>(
      hb, wq, q_bias, q_lin, M, HQ * DH, H_DIM);
  gemm_kernel<<<dim3((HKV * DH) / GBN, M / GBM), 256, GEMM_SMEM, stream>>>(
      hb, wk, k_bias, k_lin, M, HKV * DH, H_DIM);
  gemm_kernel<<<dim3((HKV * DH) / GBN, M / GBM), 256, GEMM_SMEM, stream>>>(
      hb, wv, v_bias, v_lin, M, HKV * DH, H_DIM);

  rope_q_kernel<<<M, 256, 0, stream>>>(q_lin, qbuf, S);
  rope_k_kernel<<<M, 256, 0, stream>>>(k_lin, kbuf, S);
  vtrans_kernel<<<M, 256, 0, stream>>>(v_lin, vtb, S);

  attn_kernel<<<dim3(S / 64, HQ, B), 128, ATTN_SMEM, stream>>>(
      qbuf, kbuf, vtb, attn, S);

  gemm_kernel<<<dim3(H_DIM / GBN, M / GBM), 256, GEMM_SMEM, stream>>>(
      attn, wo, nullptr, (float*)d_out, M, H_DIM, HQ * DH);
}
